// QGraphConv_19018115187414
// MI455X (gfx1250) — 8-bit matrix operand order repaired and checked by reading, compiled, not run on hardware
//
#include <hip/hip_runtime.h>
#include <math.h>

typedef __attribute__((ext_vector_type(8))) int v8i;

#define IN_F 256
#define OUT_F 64

__device__ __forceinline__ unsigned enc_f32(float f) {
  unsigned b = __float_as_uint(f);
  return (b & 0x80000000u) ? ~b : (b | 0x80000000u);
}
__device__ __forceinline__ float dec_f32(unsigned u) {
  return (u & 0x80000000u) ? __uint_as_float(u & 0x7fffffffu)
                           : __uint_as_float(~u);
}
__device__ __forceinline__ int qlvl(float v, float mn, float inv_s, float qmax) {
  float t = (v - mn) * inv_s;
  t = fminf(fmaxf(t, 0.0f), qmax);
  return (int)rintf(t);
}

__global__ void init_kernel(float* __restrict__ out, int* __restrict__ deg_out,
                            int* __restrict__ deg_in, unsigned* __restrict__ encmm,
                            int out_elems, int n_nodes) {
  int i = blockIdx.x * blockDim.x + threadIdx.x;
  if (i < out_elems) out[i] = 0.0f;
  if (i < n_nodes) { deg_out[i] = 0; deg_in[i] = 0; }
  if (i == 0) { encmm[0] = 0xFFFFFFFFu; encmm[1] = 0u; }
}

__global__ void degree_kernel(const int* __restrict__ src,
                              const int* __restrict__ dst,
                              int* __restrict__ deg_out,
                              int* __restrict__ deg_in, int n_edges) {
  int e = blockIdx.x * blockDim.x + threadIdx.x;
  if (e < n_edges) {
    atomicAdd(&deg_out[src[e]], 1);
    atomicAdd(&deg_in[dst[e]], 1);
  }
}

__global__ void rs_kernel(const int* __restrict__ deg_out,
                          const int* __restrict__ deg_in,
                          float* __restrict__ rs_out, float* __restrict__ rs_in,
                          int n_nodes) {
  int i = blockIdx.x * blockDim.x + threadIdx.x;
  if (i < n_nodes) {
    rs_out[i] = 1.0f / sqrtf(fmaxf((float)deg_out[i], 1.0f));
    rs_in[i] = 1.0f / sqrtf(fmaxf((float)deg_in[i], 1.0f));
  }
}

__global__ void wminmax_kernel(const float* __restrict__ w, int n,
                               float* __restrict__ wmm) {
  __shared__ float smin[256], smax[256];
  float mn = 3.4e38f, mx = -3.4e38f;
  for (int i = threadIdx.x; i < n; i += 256) {
    float v = w[i];
    mn = fminf(mn, v); mx = fmaxf(mx, v);
  }
  smin[threadIdx.x] = mn; smax[threadIdx.x] = mx;
  __syncthreads();
  for (int s = 128; s > 0; s >>= 1) {
    if ((int)threadIdx.x < s) {
      smin[threadIdx.x] = fminf(smin[threadIdx.x], smin[threadIdx.x + s]);
      smax[threadIdx.x] = fmaxf(smax[threadIdx.x], smax[threadIdx.x + s]);
    }
    __syncthreads();
  }
  if (threadIdx.x == 0) { wmm[0] = smin[0]; wmm[1] = smax[0]; }
}

__global__ void fminmax_kernel(const float* __restrict__ feat,
                               const float* __restrict__ rs_out,
                               unsigned* __restrict__ encmm, int total) {
  __shared__ float smin[256], smax[256];
  float mn = 3.4e38f, mx = -3.4e38f;
  for (int i = blockIdx.x * blockDim.x + threadIdx.x; i < total;
       i += gridDim.x * blockDim.x) {
    const float v = feat[i] * rs_out[i >> 8];
    mn = fminf(mn, v); mx = fmaxf(mx, v);
  }
  smin[threadIdx.x] = mn; smax[threadIdx.x] = mx;
  __syncthreads();
  for (int s = 128; s > 0; s >>= 1) {
    if ((int)threadIdx.x < s) {
      smin[threadIdx.x] = fminf(smin[threadIdx.x], smin[threadIdx.x + s]);
      smax[threadIdx.x] = fmaxf(smax[threadIdx.x], smax[threadIdx.x + s]);
    }
    __syncthreads();
  }
  if (threadIdx.x == 0) {
    atomicMin(&encmm[0], enc_f32(smin[0]));
    atomicMax(&encmm[1], enc_f32(smax[0]));
  }
}

__global__ void params_kernel(const unsigned* __restrict__ encmm,
                              const float* __restrict__ wmm,
                              const int* __restrict__ num_bits,
                              float* __restrict__ params) {
  if (threadIdx.x == 0 && blockIdx.x == 0) {
    const float fmin = dec_f32(encmm[0]);
    const float fmax = dec_f32(encmm[1]);
    const float qmax = exp2f((float)num_bits[0]) - 1.0f;
    const float sf = fmaxf((fmax - fmin) / qmax, 1e-8f);
    const float sw = fmaxf((wmm[1] - wmm[0]) / qmax, 1e-8f);
    params[0] = sf;
    params[1] = fmin;
    params[2] = sw;
    params[3] = wmm[0];
    params[4] = qmax;
    params[5] = 1.0f / sf;
    params[6] = 1.0f / sw;
  }
}

__global__ void packb_kernel(const float* __restrict__ w,
                             const float* __restrict__ params,
                             int* __restrict__ packedB,
                             int* __restrict__ colsum) {
  const float mw = params[3], qmax = params[4], inv_sw = params[6];
  for (int i = 0; i < 16; ++i) {
    const int p = (int)threadIdx.x + (i << 8);
    const int lane = p & 31, j = (p >> 5) & 7, c = (p >> 8) & 3, t = (p >> 10) & 3;
    const int hi = lane >> 4;
    const int n = (t << 4) + (lane & 15);
    int packed = 0;
    for (int b = 0; b < 4; ++b) {
      const int K = (c << 6) + ((j >> 2) << 5) + (hi << 4) + ((j & 3) << 2) + b;
      const int lvl = qlvl(w[K * OUT_F + n], mw, inv_sw, qmax);
      packed |= lvl << (8 * b);
    }
    packedB[p] = packed;
  }
  if (threadIdx.x < OUT_F) {
    int s = 0;
    for (int K = 0; K < IN_F; ++K)
      s += qlvl(w[K * OUT_F + (int)threadIdx.x], mw, inv_sw, qmax);
    colsum[threadIdx.x] = s;
  }
}

__global__ void __launch_bounds__(256) gemm_kernel(
    const float* __restrict__ feat, const float* __restrict__ rs_out,
    const float* __restrict__ params, const int* __restrict__ packedB,
    const int* __restrict__ colsum, float* __restrict__ hbuf, int n_nodes) {
  const int ntiles = (n_nodes + 15) >> 4;
  const int tile = blockIdx.x * (blockDim.x >> 5) + ((int)threadIdx.x >> 5);
  if (tile >= ntiles) return;
  const int lane = threadIdx.x & 31;
  const int hi = lane >> 4;
  const int ml = lane & 15;
  int arow = tile * 16 + ml;
  if (arow >= n_nodes) arow = n_nodes - 1;
  const float sf = params[0], mf = params[1], sw = params[2], mw = params[3];
  const float qmax = params[4], inv_sf = params[5];
  const float rs = rs_out[arow];
  const float* frow = feat + (size_t)arow * IN_F;

  v8i acc[4];
#pragma unroll
  for (int t = 0; t < 4; ++t)
#pragma unroll
    for (int r = 0; r < 8; ++r) acc[t][r] = 0;

  int rsum = 0;
#pragma unroll
  for (int c = 0; c < 4; ++c) {
    v8i a;
#pragma unroll
    for (int j = 0; j < 8; ++j) {
      const int k0 = (c << 6) + ((j >> 2) << 5) + (hi << 4) + ((j & 3) << 2);
      const float4 f = *(const float4*)(frow + k0);
      const int l0 = qlvl(f.x * rs, mf, inv_sf, qmax);
      const int l1 = qlvl(f.y * rs, mf, inv_sf, qmax);
      const int l2 = qlvl(f.z * rs, mf, inv_sf, qmax);
      const int l3 = qlvl(f.w * rs, mf, inv_sf, qmax);
      rsum += l0 + l1 + l2 + l3;
      a[j] = l0 | (l1 << 8) | (l2 << 16) | (l3 << 24);
    }
#pragma unroll
    for (int t = 0; t < 4; ++t) {
      v8i b;
      const int base = (((t << 2) + c) << 8) + lane;
#pragma unroll
      for (int j = 0; j < 8; ++j) b[j] = packedB[base + (j << 5)];
      acc[t] = __builtin_amdgcn_wmma_i32_16x16x64_iu8(false, a, false, b,
                                                      acc[t], false, false);
    }
  }
  const int rfull = rsum + __shfl_xor(rsum, 16);
  const float kterm = (float)IN_F * mf * mw;
  float rowadd[8];
#pragma unroll
  for (int r = 0; r < 8; ++r) {
    const int rsr = __shfl(rfull, r + (hi << 3));
    rowadd[r] = sf * mw * (float)rsr + kterm;
  }
#pragma unroll
  for (int t = 0; t < 4; ++t) {
    const int col = (t << 4) + ml;
    const float colterm = sw * mf * (float)colsum[col];
#pragma unroll
    for (int r = 0; r < 8; ++r) {
      const int row = tile * 16 + r + (hi << 3);
      if (row < n_nodes) {
        const float h = sf * sw * (float)acc[t][r] + rowadd[r] + colterm;
        hbuf[(size_t)row * OUT_F + col] = h;
      }
    }
  }
}

__global__ void scatter_kernel(const int* __restrict__ src,
                               const int* __restrict__ dst,
                               const float* __restrict__ hbuf,
                               float* __restrict__ out, int total) {
  const int i = blockIdx.x * blockDim.x + threadIdx.x;
  if (i >= total) return;
  const int e = i >> 6;
  const int col = i & 63;
  const float v = hbuf[(size_t)src[e] * OUT_F + col];
  float* p = out + (size_t)dst[e] * OUT_F + col;
  asm volatile("global_atomic_add_f32 %0, %1, off" : : "v"(p), "v"(v) : "memory");
}

__global__ void finalize_kernel(float* __restrict__ out,
                                const float* __restrict__ rs_in,
                                const float* __restrict__ bias, int total) {
  const int i = blockIdx.x * blockDim.x + threadIdx.x;
  if (i >= total) return;
  out[i] = out[i] * rs_in[i >> 6] + bias[i & 63];
}

extern "C" void kernel_launch(void* const* d_in, const int* in_sizes, int n_in,
                              void* d_out, int out_size, void* d_ws, size_t ws_size,
                              hipStream_t stream) {
  (void)n_in; (void)out_size; (void)ws_size;
  const float* feat = (const float*)d_in[0];
  const float* weight = (const float*)d_in[1];
  const float* bias = (const float*)d_in[2];
  const int* src = (const int*)d_in[3];
  const int* dst = (const int*)d_in[4];
  const int* num_bits = (const int*)d_in[5];
  float* out = (float*)d_out;

  const int n_nodes = in_sizes[0] / IN_F;
  const int n_edges = in_sizes[3];
  const int out_elems = n_nodes * OUT_F;

  char* ws = (char*)d_ws;
  size_t off = 0;
  float* hbuf = (float*)(ws + off);        off += (size_t)out_elems * 4;
  int* deg_out = (int*)(ws + off);         off += (size_t)n_nodes * 4;
  int* deg_in = (int*)(ws + off);          off += (size_t)n_nodes * 4;
  float* rs_out = (float*)(ws + off);      off += (size_t)n_nodes * 4;
  float* rs_in = (float*)(ws + off);       off += (size_t)n_nodes * 4;
  unsigned* encmm = (unsigned*)(ws + off); off += 16;
  float* wmm = (float*)(ws + off);         off += 16;
  float* params = (float*)(ws + off);      off += 64;
  int* colsum = (int*)(ws + off);          off += 64 * 4;
  int* packedB = (int*)(ws + off);         off += 4096 * 4;

  init_kernel<<<(out_elems + 255) / 256, 256, 0, stream>>>(
      out, deg_out, deg_in, encmm, out_elems, n_nodes);
  degree_kernel<<<(n_edges + 255) / 256, 256, 0, stream>>>(
      src, dst, deg_out, deg_in, n_edges);
  rs_kernel<<<(n_nodes + 255) / 256, 256, 0, stream>>>(
      deg_out, deg_in, rs_out, rs_in, n_nodes);
  wminmax_kernel<<<1, 256, 0, stream>>>(weight, in_sizes[1], wmm);
  fminmax_kernel<<<1024, 256, 0, stream>>>(feat, rs_out, encmm, in_sizes[0]);
  params_kernel<<<1, 1, 0, stream>>>(encmm, wmm, num_bits, params);
  packb_kernel<<<1, 256, 0, stream>>>(weight, params, packedB, colsum);
  const int ntiles = (n_nodes + 15) / 16;
  gemm_kernel<<<(ntiles + 7) / 8, 256, 0, stream>>>(
      feat, rs_out, params, packedB, colsum, hbuf, n_nodes);
  const int scat_total = n_edges * OUT_F;
  scatter_kernel<<<(scat_total + 255) / 256, 256, 0, stream>>>(
      src, dst, hbuf, out, scat_total);
  finalize_kernel<<<(out_elems + 255) / 256, 256, 0, stream>>>(
      out, rs_in, bias, out_elems);
}
